// Attention_17987323036182
// MI455X (gfx1250) — matrix operand order repaired and checked by reading, not compiled, not run on hardware
//
#include <hip/hip_runtime.h>

// ---------------- problem constants ----------------
constexpr int S_   = 2048;
constexpr int DIM_ = 2048;
constexpr int H_   = 16;
constexpr int G_   = 2;
constexpr int DH_  = 128;
constexpr int REP_ = 8;          // H/G
constexpr int L_   = 32;         // compression block length
constexpr int DSTR = 16;         // compression stride
constexpr int LP_  = 32;         // selection block size
constexpr int NSEL = 16;
constexpr int W_   = 256;        // sliding window
constexpr int J_   = 127;        // (S-L)/D + 1
constexpr int NB_  = 64;         // S/LP
constexpr float SCALE_ = 0.08838834764831845f;  // DH^-0.5

// ---------------- WMMA plumbing ----------------
typedef __bf16 bf16x16 __attribute__((ext_vector_type(16)));
typedef float  v8f     __attribute__((ext_vector_type(8)));

__device__ inline unsigned short f2bf(float f) {
  unsigned int u = __float_as_uint(f);
  u += 0x7FFFu + ((u >> 16) & 1u);        // round-to-nearest-even
  return (unsigned short)(u >> 16);
}

union FragCvt { uint4 u[2]; bf16x16 v; };

__device__ inline bf16x16 frag_ld2(const unsigned short* p0, const unsigned short* p1) {
  FragCvt f;
  f.u[0] = *reinterpret_cast<const uint4*>(p0);
  f.u[1] = *reinterpret_cast<const uint4*>(p1);
  return f.v;
}
__device__ inline bf16x16 frag_ld16(const unsigned short* p) { return frag_ld2(p, p + 8); }

__device__ inline v8f wmma_bf16(bf16x16 a, bf16x16 b, v8f c) {
  return __builtin_amdgcn_wmma_f32_16x16x32_bf16(false, a, false, b, (short)0, c, false, false);
}

__device__ inline v8f v8zero() { v8f z = {0.f,0.f,0.f,0.f,0.f,0.f,0.f,0.f}; return z; }

// async global -> LDS (16B per lane), tracked by ASYNCcnt
__device__ inline void async_ld_b128(const void* gptr, void* lds_ptr) {
  unsigned lds_off = (unsigned)(size_t)lds_ptr;   // low 32 bits of generic ptr == LDS byte offset
  asm volatile("global_load_async_to_lds_b128 %0, %1, off"
               :: "v"(lds_off), "v"(gptr) : "memory");
}
__device__ inline void async_wait0() {
  asm volatile("s_wait_asynccnt 0x0" ::: "memory");
}

__device__ inline float redmax16(float v) {   // reduce within 16-lane half (row of C tile)
  v = fmaxf(v, __shfl_xor(v, 1, 32));
  v = fmaxf(v, __shfl_xor(v, 2, 32));
  v = fmaxf(v, __shfl_xor(v, 4, 32));
  v = fmaxf(v, __shfl_xor(v, 8, 32));
  return v;
}
__device__ inline float redsum16(float v) {
  v += __shfl_xor(v, 1, 32);
  v += __shfl_xor(v, 2, 32);
  v += __shfl_xor(v, 4, 32);
  v += __shfl_xor(v, 8, 32);
  return v;
}

// ---------------- 1. fp32 -> bf16 ----------------
__global__ void f2bf_kernel(const float* __restrict__ src, unsigned short* __restrict__ dst, int n) {
  int i = blockIdx.x * blockDim.x + threadIdx.x;
  if (i < n) dst[i] = f2bf(src[i]);
}

// ---------------- 2. bf16 WMMA GEMM: C(MxN) = A(MxK) @ B(KxN), A/B bf16 row-major, C fp32 ----
constexpr int BM = 64, BN = 128, BK = 32;

__global__ __launch_bounds__(256) void gemm_bf16(const unsigned short* __restrict__ A,
                                                 const unsigned short* __restrict__ B,
                                                 float* __restrict__ C, int M, int N, int K)
{
  __shared__ __align__(16) unsigned short Ab[BM * BK];   // [row][k]
  __shared__ __align__(16) unsigned short Bt[BN * BK];   // [n][k] (transposed)
  const int tid  = threadIdx.x;
  const int wave = tid >> 5;
  const int lane = tid & 31;
  const int hf   = lane >> 4;
  const int l15  = lane & 15;
  const int bm = blockIdx.y * BM, bn = blockIdx.x * BN;
  const int wm = (wave & 1) * 32, wn = (wave >> 1) * 32;

  v8f acc[2][2];
#pragma unroll
  for (int i = 0; i < 2; ++i)
#pragma unroll
    for (int j = 0; j < 2; ++j) acc[i][j] = v8zero();

  const int arow = tid >> 2, ach = (tid & 3) * 8;

  for (int k0 = 0; k0 < K; k0 += BK) {
    __syncthreads();
    // A tile: 64x32 bf16, async global->LDS, 16B per thread
    async_ld_b128(A + (size_t)(bm + arow) * K + k0 + ach, Ab + arow * BK + ach);
    // B tile: 32x128 bf16, transposed into Bt[n][k]
#pragma unroll
    for (int i = 0; i < 2; ++i) {
      int idx = tid * 2 + i;
      int kr = idx >> 4, ch = (idx & 15) * 8;
      uint4 dv = *reinterpret_cast<const uint4*>(B + (size_t)(k0 + kr) * N + bn + ch);
      const unsigned short* dsv = reinterpret_cast<const unsigned short*>(&dv);
#pragma unroll
      for (int jj = 0; jj < 8; ++jj) Bt[(ch + jj) * BK + kr] = dsv[jj];
    }
    if (k0 + BK < K) {
      __builtin_prefetch(A + (size_t)(bm + arow) * K + k0 + BK + ach, 0, 1);
      __builtin_prefetch(B + (size_t)(k0 + BK + (tid >> 4)) * N + bn, 0, 1);
    }
    async_wait0();
    __syncthreads();

    bf16x16 af[2], bfg[2];
#pragma unroll
    for (int i = 0; i < 2; ++i) {
      // A frag 16x32: lane<16 -> K[0..8)+K[16..24), lane>=16 -> K[8..16)+K[24..32)
      const unsigned short* p = Ab + (wm + i * 16 + l15) * BK + (hf ? 8 : 0);
      af[i] = frag_ld2(p, p + 16);
    }
#pragma unroll
    for (int j = 0; j < 2; ++j) {
      // B frag 32x16: lane<16 -> col l15, K[0..8)+K[16..24) ; lane>=16 -> K[8..16)+K[24..32)
      const unsigned short* p = Bt + (wn + j * 16 + l15) * BK + (hf ? 8 : 0);
      bfg[j] = frag_ld2(p, p + 16);
    }
#pragma unroll
    for (int i = 0; i < 2; ++i)
#pragma unroll
      for (int j = 0; j < 2; ++j)
        acc[i][j] = wmma_bf16(af[i], bfg[j], acc[i][j]);
  }
  // C store: VGPR e -> row e (+8 for upper half)
#pragma unroll
  for (int i = 0; i < 2; ++i)
#pragma unroll
    for (int j = 0; j < 2; ++j) {
      int col  = bn + wn + j * 16 + l15;
      int row0 = bm + wm + i * 16 + hf * 8;
#pragma unroll
      for (int e = 0; e < 8; ++e)
        C[(size_t)(row0 + e) * N + col] = acc[i][j][e];
    }
}

// ---------------- 3. RoPE (interleaved) + bf16 mirror ----------------
__global__ void rope_conv(float* __restrict__ t, unsigned short* __restrict__ tb,
                          const float* __restrict__ cosp, const float* __restrict__ sinp,
                          int heads, int doRope)
{
  int idx = blockIdx.x * blockDim.x + threadIdx.x;
  int total = S_ * heads * (DH_ / 2);
  if (idx >= total) return;
  int i   = idx & 63;
  int rem = idx >> 6;
  int hh  = rem % heads;
  int s   = rem / heads;
  size_t base = ((size_t)s * heads + hh) * DH_ + 2 * i;
  float x0 = t[base], x1 = t[base + 1];
  if (doRope) {
    float c = cosp[s * 64 + i], sn = sinp[s * 64 + i];
    float y0 = x0 * c - x1 * sn;
    float y1 = x0 * sn + x1 * c;
    x0 = y0; x1 = y1;
    t[base] = x0; t[base + 1] = x1;
  }
  tb[base]     = f2bf(x0);
  tb[base + 1] = f2bf(x1);
}

// ---------------- 4. compressed K/V means ----------------
__global__ void cmp_kv(const float* __restrict__ kf, const float* __restrict__ vf,
                       float* __restrict__ kcmp, float* __restrict__ vcmp)
{
  int idx = blockIdx.x * blockDim.x + threadIdx.x;   // J*G*DH
  if (idx >= J_ * G_ * DH_) return;
  int d = idx & 127, g = (idx >> 7) & 1, j = idx >> 8;
  int t0 = j * DSTR;
  float ak = 0.f, av = 0.f;
  for (int i = 0; i < L_; ++i) {
    size_t o = ((size_t)(t0 + i) * G_ + g) * DH_ + d;
    ak += kf[o];
    av += vf[o];
  }
  kcmp[idx] = ak * (1.0f / L_);
  vcmp[idx] = av * (1.0f / L_);
}

// ---------------- 5. compressed attention + importance + top-k + gates ----------------
__global__ __launch_bounds__(128) void cmp_attn(const float* __restrict__ qf,
                                                const float* __restrict__ kcmp,
                                                const float* __restrict__ vcmp,
                                                const float* __restrict__ wg,
                                                float* __restrict__ ocmp,
                                                float* __restrict__ gates,
                                                unsigned long long* __restrict__ bmask)
{
  const int s = blockIdx.x, g = blockIdx.y, tid = threadIdx.x;
  __shared__ float qs8[REP_ * DH_];
  __shared__ float scb[REP_ * J_];
  __shared__ float ps[J_];
  __shared__ float impv[NB_];

  for (int i = tid; i < REP_ * DH_; i += 128) {
    int r = i >> 7, d = i & 127;
    qs8[i] = qf[((size_t)s * H_ + g * REP_ + r) * DH_ + d];
  }
  __syncthreads();

  const int nvalid = (s >= L_ - 1) ? ((s - (L_ - 1)) >> 4) + 1 : 0;  // j*16+31 <= s

  for (int idx = tid; idx < REP_ * J_; idx += 128) {
    int r = idx / J_, j = idx % J_;
    float a = -1e30f;
    if (j < nvalid) {
      const float* kp = kcmp + ((size_t)j * G_ + g) * DH_;
      float acc = 0.f;
      for (int d = 0; d < DH_; ++d) acc += qs8[r * DH_ + d] * kp[d];
      a = acc * SCALE_;
    }
    scb[idx] = a;
  }
  __syncthreads();

  if (tid < REP_) {
    float* row = scb + tid * J_;
    if (nvalid == 0) {
      for (int j = 0; j < J_; ++j) row[j] = 0.f;
    } else {
      float m = -1e30f;
      for (int j = 0; j < nvalid; ++j) m = fmaxf(m, row[j]);
      float l = 0.f;
      for (int j = 0; j < nvalid; ++j) { float p = __expf(row[j] - m); row[j] = p; l += p; }
      float inv = 1.f / l;
      for (int j = 0; j < nvalid; ++j) row[j] *= inv;
      for (int j = nvalid; j < J_; ++j) row[j] = 0.f;
    }
  }
  __syncthreads();

  for (int idx = tid; idx < REP_ * DH_; idx += 128) {
    int r = idx >> 7, d = idx & 127;
    float acc = 0.f;
    for (int j = 0; j < nvalid; ++j)
      acc += scb[r * J_ + j] * vcmp[((size_t)j * G_ + g) * DH_ + d];
    ocmp[((size_t)s * H_ + g * REP_ + r) * DH_ + d] = acc;
  }

  if (tid < J_) {
    float a = 0.f;
    for (int r = 0; r < REP_; ++r) a += scb[r * J_ + tid];
    ps[tid] = a;
  }
  __syncthreads();

  if (tid < NB_) {
    int m = tid;
    float v = -1e30f;
    if (m * LP_ <= s) {              // causal at block level
      v = ps[2 * m];                 // 2m <= 126 always
      if (m > 0)          v += ps[2 * m - 1];
      if (2 * m + 1 < J_) v += ps[2 * m + 1];
      int cur = s / LP_;
      if (m == cur) v += 1e9f;
      if (m == 0)   v += 1e9f;
    }
    impv[m] = v;
  }
  __syncthreads();

  if (tid == 0) {
    unsigned long long bits = 0;
    for (int it = 0; it < NSEL; ++it) {
      float best = -1e29f; int bi = -1;
      for (int m = 0; m < NB_; ++m) {
        if ((bits >> m) & 1ull) continue;
        if (impv[m] > best) { best = impv[m]; bi = m; }
      }
      if (bi < 0) break;
      bits |= 1ull << bi;
    }
    bmask[(size_t)s * G_ + g] = bits;
  }

  if (tid < REP_ * 3) {
    int r = tid / 3, e = tid % 3;
    float a = 0.f;
    for (int d = 0; d < DH_; ++d) a += qs8[r * DH_ + d] * wg[d * 3 + e];
    gates[(((size_t)s * G_ + g) * REP_ + r) * 3 + e] = 1.f / (1.f + __expf(-a));
  }
}

// ---------------- 6. WMMA flash attention (mode0 = selected blocks, mode1 = sliding window) --
__global__ __launch_bounds__(256) void sel_sw_attn(
    const unsigned short* __restrict__ qb,    // (S,H,DH) bf16, roped
    const unsigned short* __restrict__ kbf,   // (S,G,DH) bf16, roped
    const unsigned short* __restrict__ vbf,   // (S,G,DH) bf16
    const float* __restrict__ gates,          // (S,G,REP,3)
    const unsigned long long* __restrict__ bmask,  // (S,G)
    const float* __restrict__ ocmp,           // (S,H,DH)     (mode0)
    float* __restrict__ partial,              // (S,H,DH) fp32: w in mode0, r in mode1
    unsigned short* __restrict__ attn_bf,     // (S,H,DH) bf16 (mode1)
    int mode)
{
  __shared__ __align__(16) unsigned short Kt[32 * 128];  // [key][dim]
  __shared__ __align__(16) unsigned short Vt[128 * 32];  // [dim][key]
  __shared__ __align__(16) unsigned short Pt[8][16 * 32];
  __shared__ unsigned long long uM;

  const int q0   = blockIdx.x * 16;
  const int g    = blockIdx.y;
  const int tid  = threadIdx.x;
  const int wave = tid >> 5;        // head within group
  const int lane = tid & 31;
  const int hf   = lane >> 4;
  const int l15  = lane & 15;
  const int h    = g * REP_ + wave;

  if (tid == 0) {
    unsigned long long u = 0;
    for (int i = 0; i < 16; ++i) u |= bmask[(size_t)(q0 + i) * G_ + g];
    uM = u;
  }

  unsigned long long qm[8];
  int qsr[8];
  float ga[8], gb[8];
#pragma unroll
  for (int e = 0; e < 8; ++e) {
    qsr[e] = q0 + e + hf * 8;
    qm[e]  = bmask[(size_t)qsr[e] * G_ + g];
    size_t gbase = (((size_t)qsr[e] * G_ + g) * REP_ + wave) * 3;
    if (mode == 0) { ga[e] = gates[gbase + 0]; gb[e] = gates[gbase + 1]; }
    else           { ga[e] = gates[gbase + 2]; gb[e] = 0.f; }
  }

  bf16x16 qfr[4];
  {
    const unsigned short* base = qb + ((size_t)(q0 + l15) * H_ + h) * DH_ + (hf ? 8 : 0);
#pragma unroll
    for (int kk = 0; kk < 4; ++kk)
      qfr[kk] = frag_ld2(base + kk * 32, base + kk * 32 + 16);
  }
  __syncthreads();

  v8f acc[8];
#pragma unroll
  for (int nt = 0; nt < 8; ++nt) acc[nt] = v8zero();
  float mst[8], lst[8];
#pragma unroll
  for (int e = 0; e < 8; ++e) { mst[e] = -1e30f; lst[e] = 0.f; }

  unsigned long long list = uM;
  int lo = q0 - (W_ - 1); if (lo < 0) lo = 0;
  int bcur = lo >> 5;
  const int bhi = (q0 + 15) >> 5;

  while (true) {
    int b;
    if (mode == 0) {
      if (list == 0) break;
      b = __builtin_ctzll(list);
      list &= list - 1;
    } else {
      if (bcur > bhi) break;
      b = bcur++;
    }
    const int t0 = b * 32;

    __syncthreads();
    {   // cooperative K/V tile load: 32 tokens x 128 dims
      const int row = tid >> 3, part = (tid & 7) * 16;
      const unsigned short* ks = kbf + ((size_t)(t0 + row) * G_ + g) * DH_ + part;
      const unsigned short* vs = vbf + ((size_t)(t0 + row) * G_ + g) * DH_ + part;
      // K tile: async global -> LDS (no transpose needed)
      async_ld_b128(ks,     Kt + row * 128 + part);
      async_ld_b128(ks + 8, Kt + row * 128 + part + 8);
      // V tile: transpose in-register, Vt[dim][key]
      uint4 v0v = *reinterpret_cast<const uint4*>(vs);
      uint4 v1v = *reinterpret_cast<const uint4*>(vs + 8);
      const unsigned short* vv0 = reinterpret_cast<const unsigned short*>(&v0v);
      const unsigned short* vv1 = reinterpret_cast<const unsigned short*>(&v1v);
#pragma unroll
      for (int jj = 0; jj < 8; ++jj) {
        Vt[(part + jj)     * 32 + row] = vv0[jj];
        Vt[(part + 8 + jj) * 32 + row] = vv1[jj];
      }
      async_wait0();
    }
    __syncthreads();

    // scores: two 16x16 tiles over 32 keys
    v8f sc0 = v8zero(), sc1 = v8zero();
#pragma unroll
    for (int kk = 0; kk < 4; ++kk) {
      const unsigned short* p0 = Kt + (l15)      * 128 + kk * 32 + (hf ? 8 : 0);
      const unsigned short* p1 = Kt + (16 + l15) * 128 + kk * 32 + (hf ? 8 : 0);
      sc0 = wmma_bf16(qfr[kk], frag_ld2(p0, p0 + 16), sc0);
      sc1 = wmma_bf16(qfr[kk], frag_ld2(p1, p1 + 16), sc1);
    }

    const int tcol0 = t0 + l15, tcol1 = t0 + 16 + l15;
    float v0a[8], v1a[8], scal[8], mnew[8];
#pragma unroll
    for (int e = 0; e < 8; ++e) {
      float s0 = sc0[e] * SCALE_;
      float s1 = sc1[e] * SCALE_;
      bool ok0, ok1;
      if (mode == 0) {
        bool bb = (qm[e] >> b) & 1ull;
        ok0 = bb && (tcol0 <= qsr[e]);
        ok1 = bb && (tcol1 <= qsr[e]);
      } else {
        ok0 = (tcol0 <= qsr[e]) && (tcol0 > qsr[e] - W_);
        ok1 = (tcol1 <= qsr[e]) && (tcol1 > qsr[e] - W_);
      }
      v0a[e] = ok0 ? s0 : -1e30f;
      v1a[e] = ok1 ? s1 : -1e30f;
      float rm = redmax16(fmaxf(v0a[e], v1a[e]));
      float mn = fmaxf(mst[e], rm);
      mnew[e] = mn;
      scal[e] = __expf(mst[e] - mn);
      mst[e]  = mn;
    }
    float p0a[8], p1a[8];
#pragma unroll
    for (int e = 0; e < 8; ++e) {
      p0a[e] = (v0a[e] > -1e29f) ? __expf(v0a[e] - mnew[e]) : 0.f;
      p1a[e] = (v1a[e] > -1e29f) ? __expf(v1a[e] - mnew[e]) : 0.f;
      lst[e] = lst[e] * scal[e] + redsum16(p0a[e] + p1a[e]);
    }
#pragma unroll
    for (int nt = 0; nt < 8; ++nt)
#pragma unroll
      for (int e = 0; e < 8; ++e)
        acc[nt][e] *= scal[e];

    // P (C-layout) -> LDS -> A-fragment layout
    unsigned short* pw = &Pt[wave][0];
#pragma unroll
    for (int e = 0; e < 8; ++e) {
      int row = e + hf * 8;
      pw[row * 32 + l15]      = f2bf(p0a[e]);
      pw[row * 32 + 16 + l15] = f2bf(p1a[e]);
    }
    const unsigned short* pr = pw + l15 * 32 + (hf ? 8 : 0);
    bf16x16 pf = frag_ld2(pr, pr + 16);

#pragma unroll
    for (int nt = 0; nt < 8; ++nt) {
      const unsigned short* vp = Vt + (nt * 16 + l15) * 32 + (hf ? 8 : 0);
      acc[nt] = wmma_bf16(pf, frag_ld2(vp, vp + 16), acc[nt]);
    }
  }

  // finalize
#pragma unroll
  for (int nt = 0; nt < 8; ++nt)
#pragma unroll
    for (int e = 0; e < 8; ++e) {
      float o = acc[nt][e] / fmaxf(lst[e], 1e-30f);
      size_t oidx = ((size_t)qsr[e] * H_ + h) * DH_ + nt * 16 + l15;
      if (mode == 0)
        partial[oidx] = ga[e] * ocmp[oidx] + gb[e] * o;   // g0*o_cmp + g1*o_sel
      else
        attn_bf[oidx] = f2bf(partial[oidx] + ga[e] * o);  // + g2*o_sw
    }
}

// ---------------- host-side orchestration ----------------
extern "C" void kernel_launch(void* const* d_in, const int* in_sizes, int n_in,
                              void* d_out, int out_size, void* d_ws, size_t ws_size,
                              hipStream_t stream)
{
  (void)in_sizes; (void)n_in; (void)out_size; (void)ws_size;
  const float* x    = (const float*)d_in[0];
  const float* fcos = (const float*)d_in[1];
  const float* fsin = (const float*)d_in[2];
  const float* wq   = (const float*)d_in[3];
  const float* wk   = (const float*)d_in[4];
  const float* wv   = (const float*)d_in[5];
  const float* wo   = (const float*)d_in[6];
  const float* wg   = (const float*)d_in[7];
  float* outp = (float*)d_out;

  char* w = (char*)d_ws;
  size_t off = 0;
  auto alloc = [&](size_t bytes) -> char* {
    char* p = w + off;
    off += (bytes + 255) & ~(size_t)255;
    return p;
  };

  unsigned short* x_bf   = (unsigned short*)alloc((size_t)S_ * DIM_ * 2);
  unsigned short* wq_bf  = (unsigned short*)alloc((size_t)DIM_ * H_ * DH_ * 2);
  unsigned short* wk_bf  = (unsigned short*)alloc((size_t)DIM_ * G_ * DH_ * 2);
  unsigned short* wv_bf  = (unsigned short*)alloc((size_t)DIM_ * G_ * DH_ * 2);
  unsigned short* wo_bf  = (unsigned short*)alloc((size_t)H_ * DH_ * DIM_ * 2);
  float*          qf     = (float*)alloc((size_t)S_ * H_ * DH_ * 4);   // reused as 'partial'
  float*          kf     = (float*)alloc((size_t)S_ * G_ * DH_ * 4);
  float*          vf     = (float*)alloc((size_t)S_ * G_ * DH_ * 4);
  unsigned short* q_bf   = (unsigned short*)alloc((size_t)S_ * H_ * DH_ * 2);
  unsigned short* k_bf   = (unsigned short*)alloc((size_t)S_ * G_ * DH_ * 2);
  unsigned short* v_bf   = (unsigned short*)alloc((size_t)S_ * G_ * DH_ * 2);
  float*          kcmp   = (float*)alloc((size_t)J_ * G_ * DH_ * 4);
  float*          vcmp   = (float*)alloc((size_t)J_ * G_ * DH_ * 4);
  float*          ocmp   = (float*)alloc((size_t)S_ * H_ * DH_ * 4);
  float*          gatesb = (float*)alloc((size_t)S_ * G_ * REP_ * 3 * 4);
  unsigned long long* bm = (unsigned long long*)alloc((size_t)S_ * G_ * 8);
  unsigned short* attnbf = (unsigned short*)alloc((size_t)S_ * H_ * DH_ * 2);

  auto cdiv = [](size_t a, size_t b) { return (unsigned int)((a + b - 1) / b); };

  // 1. convert inputs/weights to bf16
  f2bf_kernel<<<cdiv((size_t)S_ * DIM_, 256), 256, 0, stream>>>(x,  x_bf,  S_ * DIM_);
  f2bf_kernel<<<cdiv((size_t)DIM_ * H_ * DH_, 256), 256, 0, stream>>>(wq, wq_bf, DIM_ * H_ * DH_);
  f2bf_kernel<<<cdiv((size_t)DIM_ * G_ * DH_, 256), 256, 0, stream>>>(wk, wk_bf, DIM_ * G_ * DH_);
  f2bf_kernel<<<cdiv((size_t)DIM_ * G_ * DH_, 256), 256, 0, stream>>>(wv, wv_bf, DIM_ * G_ * DH_);
  f2bf_kernel<<<cdiv((size_t)H_ * DH_ * DIM_, 256), 256, 0, stream>>>(wo, wo_bf, H_ * DH_ * DIM_);

  // 2. projections (WMMA GEMMs)
  gemm_bf16<<<dim3((H_ * DH_) / BN, S_ / BM), 256, 0, stream>>>(x_bf, wq_bf, qf, S_, H_ * DH_, DIM_);
  gemm_bf16<<<dim3((G_ * DH_) / BN, S_ / BM), 256, 0, stream>>>(x_bf, wk_bf, kf, S_, G_ * DH_, DIM_);
  gemm_bf16<<<dim3((G_ * DH_) / BN, S_ / BM), 256, 0, stream>>>(x_bf, wv_bf, vf, S_, G_ * DH_, DIM_);

  // 3. RoPE + bf16 mirrors
  rope_conv<<<cdiv((size_t)S_ * H_ * 64, 256), 256, 0, stream>>>(qf, q_bf, fcos, fsin, H_, 1);
  rope_conv<<<cdiv((size_t)S_ * G_ * 64, 256), 256, 0, stream>>>(kf, k_bf, fcos, fsin, G_, 1);
  rope_conv<<<cdiv((size_t)S_ * G_ * 64, 256), 256, 0, stream>>>(vf, v_bf, fcos, fsin, G_, 0);

  // 4. compressed K/V
  cmp_kv<<<cdiv((size_t)J_ * G_ * DH_, 256), 256, 0, stream>>>(kf, vf, kcmp, vcmp);

  // 5. compressed attention, block selection, gates
  cmp_attn<<<dim3(S_, G_), 128, 0, stream>>>(qf, kcmp, vcmp, wg, ocmp, gatesb, bm);

  // 6. selected-block + sliding-window attention (WMMA flash, two passes)
  sel_sw_attn<<<dim3(S_ / 16, G_), 256, 0, stream>>>(q_bf, k_bf, v_bf, gatesb, bm, ocmp, qf, attnbf, 0);
  sel_sw_attn<<<dim3(S_ / 16, G_), 256, 0, stream>>>(q_bf, k_bf, v_bf, gatesb, bm, ocmp, qf, attnbf, 1);

  // 7. output projection
  gemm_bf16<<<dim3(DIM_ / BN, S_ / BM), 256, 0, stream>>>(attnbf, wo_bf, outp, S_, DIM_, H_ * DH_);
}
